// MapFormerWM_35141422416173
// MI455X (gfx1250) — matrix operand order repaired and checked by reading, not compiled, not run on hardware
//
#include <hip/hip_runtime.h>
#include <math.h>

// ---------------- constants (match reference) ----------------
constexpr int Bc  = 2;
constexpr int Tc  = 1024;
constexpr int Dc  = 1024;
constexpr int Hc  = 16;
constexpr int DHc = 64;
constexpr int Rc  = 32;
constexpr int FFc = 4096;
constexpr int OVc = 16384;
constexpr int Lc  = 2;

typedef _Float16 v16h __attribute__((ext_vector_type(16)));
typedef _Float16 h8v  __attribute__((ext_vector_type(8)));
typedef _Float16 h2v  __attribute__((ext_vector_type(2)));
typedef float    v8f  __attribute__((ext_vector_type(8)));

// ---------------- CDNA5 async global->LDS copy (ASYNCcnt-tracked, bypasses VGPRs) ----------------
__device__ __forceinline__ unsigned lds_offset_of(const void* p) {
  return (unsigned)(size_t)(__attribute__((address_space(3))) const void*)p;
}

// copy 16 bytes global -> LDS for this lane
__device__ __forceinline__ void async_copy_b128(const _Float16* gsrc, _Float16* ldst) {
  unsigned lofs = lds_offset_of(ldst);
  asm volatile("global_load_async_to_lds_b128 %0, %1, off"
               :
               : "v"(lofs), "v"(gsrc)
               : "memory");
}

__device__ __forceinline__ void wait_async0() {
  asm volatile("s_wait_asynccnt 0x0" ::: "memory");
}

// ---------------- WMMA fragment loaders ----------------
// A 16x32 (f16), row-major LDS tile, leading dim `ld` (multiple of 8):
__device__ __forceinline__ v16h lds_a_frag(const _Float16* s, int ld) {
  int lane = threadIdx.x & 31;
  const _Float16* p = s + (lane & 15) * ld + (lane >> 4) * 16;
  union { v16h v; h8v h[2]; } u;
  u.h[0] = *(const h8v*)p;
  u.h[1] = *(const h8v*)(p + 8);
  return u.v;
}

// B 32x16 (f16) pre-swizzled fragment block (512 halves): lane reads 32B contiguous.
__device__ __forceinline__ v16h lds_bfrag(const _Float16* blk) {
  int lane = threadIdx.x & 31;
  const _Float16* p = blk + lane * 16;
  union { v16h v; h8v h[2]; } u;
  u.h[0] = *(const h8v*)p;
  u.h[1] = *(const h8v*)(p + 8);
  return u.v;
}

// ---------------- weight pre-swizzle: f32 [K][N] -> f16 frag-panel layout ----------------
__global__ __launch_bounds__(256) void swz_kernel(const float* __restrict__ W,
                                                  _Float16* __restrict__ dst,
                                                  int N, int K) {
  int p  = blockIdx.x;
  int kb = blockIdx.y;
  int tid = threadIdx.x;
  int kp = tid >> 4, k_ = 2 * kp;       // K pair
  int bn = (tid & 15) * 8;              // 8-wide N group
  int n0 = p * 128, k0 = kb * 32;
  const float* wg0 = W + (size_t)(k0 + k_) * N + n0 + bn;
  const float* wg1 = wg0 + N;
  _Float16* base = dst + ((size_t)p * (K >> 5) + kb) * 4096;
#pragma unroll
  for (int j = 0; j < 8; ++j) {
    int n_ = bn + j;
    h2v t; t[0] = (_Float16)wg0[j]; t[1] = (_Float16)wg1[j];
    int addr = (((n_ >> 4) * 32) + ((k_ >> 4) * 16) + (n_ & 15)) * 16 + (k_ & 15);
    *(h2v*)(&base[addr]) = t;
  }
}

// ---------------- small kernels ----------------

__global__ __launch_bounds__(128) void delta_kernel(
    const int* __restrict__ actions, const float* __restrict__ aemb,
    const float* __restrict__ lw, const float* __restrict__ lb,
    float* __restrict__ delta) {
  __shared__ float a[Dc];
  int token = blockIdx.x;
  int tid = threadIdx.x;
  int act = actions[token];
  for (int i = tid; i < Dc; i += 128) a[i] = aemb[(size_t)act * Dc + i];
  __syncthreads();
  if (tid < Rc) {
    float acc = lb[tid];
    for (int k = 0; k < Dc; ++k) acc += a[k] * lw[k * Rc + tid];
    delta[(size_t)token * Rc + tid] = acc;
  }
}

// Prefix product of block-diag 2x2 rotations == cumulative angle sum.
__global__ void scan_kernel(const float* __restrict__ delta,
                            float* __restrict__ cosT, float* __restrict__ sinT) {
  int tid = threadIdx.x;
  if (tid >= Bc * Rc) return;
  int b = tid / Rc, r = tid % Rc;
  float acc = 0.f;
  for (int t = 0; t < Tc; ++t) {
    size_t idx = ((size_t)(b * Tc + t)) * Rc + r;
    acc += delta[idx];
    cosT[idx] = cosf(acc);
    sinT[idx] = sinf(acc);
  }
}

__global__ void gather_obs_kernel(const int* __restrict__ obs,
                                  const float* __restrict__ emb,
                                  float* __restrict__ x) {
  size_t i = (size_t)blockIdx.x * 256 + threadIdx.x;
  size_t token = i >> 10;
  size_t d = i & (Dc - 1);
  x[i] = emb[(size_t)obs[token] * Dc + d];
}

// LayerNorm over last dim (1024), one block per token; f32 in, f16 out.
__global__ __launch_bounds__(256) void ln_kernel(
    const float* __restrict__ x, const float* __restrict__ g,
    const float* __restrict__ b, _Float16* __restrict__ h) {
  int token = blockIdx.x, tid = threadIdx.x;
  const float* xr = x + (size_t)token * Dc;
  int d0 = tid * 4;
  float vals[4], s = 0.f, ss = 0.f;
#pragma unroll
  for (int i = 0; i < 4; ++i) {
    float v = xr[d0 + i];
    vals[i] = v; s += v; ss += v * v;
  }
  __shared__ float red[256], red2[256];
  red[tid] = s; red2[tid] = ss;
  __syncthreads();
  for (int off = 128; off > 0; off >>= 1) {
    if (tid < off) { red[tid] += red[tid + off]; red2[tid] += red2[tid + off]; }
    __syncthreads();
  }
  float mean = red[0] * (1.f / Dc);
  float var  = red2[0] * (1.f / Dc) - mean * mean;
  float rstd = rsqrtf(var + 1e-5f);
  _Float16* hr = h + (size_t)token * Dc;
#pragma unroll
  for (int i = 0; i < 4; ++i) {
    int d = d0 + i;
    hr[d] = (_Float16)((vals[i] - mean) * rstd * g[d] + b[d]);
  }
}

// Apply cumulative rotation to f16 Q or K in place (packed pair per thread).
__global__ void rope_kernel(_Float16* __restrict__ q,
                            const float* __restrict__ cosT,
                            const float* __restrict__ sinT) {
  size_t i = (size_t)blockIdx.x * 256 + threadIdx.x;
  int r = (int)(i & (Rc - 1));
  size_t th = i >> 5;            // (b*T + t)*H + h
  size_t bt = th >> 4;           // b*T + t
  float c = cosT[bt * Rc + r], s = sinT[bt * Rc + r];
  size_t base = th * DHc + 2 * r;
  h2v t = *(h2v*)(&q[base]);
  float q0 = (float)t[0], q1 = (float)t[1];
  t[0] = (_Float16)(c * q0 - s * q1);
  t[1] = (_Float16)(s * q0 + c * q1);
  *(h2v*)(&q[base]) = t;
}

// ---------------- WMMA GEMM, async double-buffered ----------------
// EPI: 0 = bias, 1 = bias + f32 residual add, 2 = bias + exact GELU. TOUT: float or _Float16.
template <int EPI, typename TOUT>
__global__ __launch_bounds__(256) void gemm_wmma_kernel(
    const _Float16* __restrict__ A, const _Float16* __restrict__ Wsw,
    const float* __restrict__ bias, const float* __restrict__ res,
    TOUT* __restrict__ C, int M, int N, int K) {
  constexpr int BM = 128, BK = 32, LDA = 48;
  __shared__ alignas(16) _Float16 As[2][BM * LDA];   // double-buffered A tiles
  __shared__ alignas(16) _Float16 Bs[2][8 * 512];    // double-buffered B frag panels
  int tid = threadIdx.x;
  int m0 = blockIdx.y * BM, n0 = blockIdx.x * 128;
  int w = tid >> 5, lane = tid & 31;
  int wm = w >> 1, wn = w & 1;      // 4x2 wave grid, wave tile 32x64

  v8f zero = {0.f, 0.f, 0.f, 0.f, 0.f, 0.f, 0.f, 0.f};
  v8f acc[2][4];
#pragma unroll
  for (int i = 0; i < 2; ++i)
#pragma unroll
    for (int j = 0; j < 4; ++j) acc[i][j] = zero;

  int arow = tid >> 1, acol = (tid & 1) * 16;
  const int nkb = K >> 5;
  const _Float16* wpanel = Wsw + (size_t)blockIdx.x * nkb * 4096;

  // async stage of tile kb into buffer buf (4x b128 per thread)
  auto stage = [&](int kb, int buf) {
    const _Float16* ag = A + (size_t)(m0 + arow) * K + kb * BK + acol;
    async_copy_b128(ag,     &As[buf][arow * LDA + acol]);
    async_copy_b128(ag + 8, &As[buf][arow * LDA + acol + 8]);
    const _Float16* bg = wpanel + (size_t)kb * 4096 + tid * 16;
    async_copy_b128(bg,     &Bs[buf][tid * 16]);
    async_copy_b128(bg + 8, &Bs[buf][tid * 16 + 8]);
  };

  stage(0, 0);
  for (int kb = 0; kb < nkb; ++kb) {
    int buf = kb & 1;
    wait_async0();        // own async writes for tile kb complete
    __syncthreads();      // all waves' writes visible; previous tile reads done
    if (kb + 1 < nkb) stage(kb + 1, buf ^ 1);

    v16h af[2], bf[4];
#pragma unroll
    for (int mi = 0; mi < 2; ++mi)
      af[mi] = lds_a_frag(&As[buf][(wm * 32 + mi * 16) * LDA], LDA);
#pragma unroll
    for (int ni = 0; ni < 4; ++ni)
      bf[ni] = lds_bfrag(&Bs[buf][(wn * 4 + ni) * 512]);
#pragma unroll
    for (int mi = 0; mi < 2; ++mi)
#pragma unroll
      for (int ni = 0; ni < 4; ++ni)
        acc[mi][ni] = __builtin_amdgcn_wmma_f32_16x16x32_f16(
            false, af[mi], false, bf[ni], (short)0, acc[mi][ni], false, false);
  }

  int hi = lane >> 4, col = lane & 15;
#pragma unroll
  for (int mi = 0; mi < 2; ++mi)
#pragma unroll
    for (int ni = 0; ni < 4; ++ni)
#pragma unroll
      for (int r = 0; r < 8; ++r) {
        int grow = m0 + wm * 32 + mi * 16 + r + hi * 8;
        int gcol = n0 + wn * 64 + ni * 16 + col;
        size_t idx = (size_t)grow * N + gcol;
        float v = acc[mi][ni][r] + bias[gcol];
        if (EPI == 1) v += res[idx];
        if (EPI == 2) v = 0.5f * v * (1.f + erff(v * 0.70710678118654752f));
        C[idx] = (TOUT)v;
      }
}

// ---------------- flash attention (per b,h, 128-query tile), all-f16 operands ----------------
__global__ __launch_bounds__(256) void attn_kernel(
    const _Float16* __restrict__ Q, const _Float16* __restrict__ Kr,
    const _Float16* __restrict__ V, _Float16* __restrict__ O) {
  constexpr int LQ = 72;
  __shared__ alignas(16) _Float16 Qs[128 * LQ];   // A-layout Q tile
  __shared__ alignas(16) _Float16 Ks[8 * 512];    // K^T 64(k=d) x 64(n=key) frag blocks
  __shared__ alignas(16) _Float16 Vs[8 * 512];    // V   64(k=key) x 64(n=d) frag blocks
  __shared__ alignas(16) _Float16 Ps[8 * 16 * LQ];

  int tid = threadIdx.x;
  int qt = blockIdx.x & 7;           // T/128 = 8 tiles
  int bh = blockIdx.x >> 3;
  int hd = bh & (Hc - 1);
  int b  = bh >> 4;
  int w = tid >> 5, lane = tid & 31;
  int hi = lane >> 4, lcol = lane & 15;

  { // stage 128x64 Q tile via async global->LDS (4x b128 per thread)
    int qrow = tid >> 1, dbase = (tid & 1) * 32;
    const _Float16* src =
        Q + ((size_t)(b * Tc + qt * 128 + qrow) * Hc + hd) * DHc + dbase;
#pragma unroll
    for (int v4 = 0; v4 < 4; ++v4)
      async_copy_b128(src + v4 * 8, &Qs[qrow * LQ + dbase + v4 * 8]);
    wait_async0();
  }
  __syncthreads();

  v16h qa0 = lds_a_frag(&Qs[(w * 16) * LQ + 0], LQ);
  v16h qa1 = lds_a_frag(&Qs[(w * 16) * LQ + 32], LQ);

  v8f zero = {0.f, 0.f, 0.f, 0.f, 0.f, 0.f, 0.f, 0.f};
  v8f ov[4];
#pragma unroll
  for (int j = 0; j < 4; ++j) ov[j] = zero;
  float mrun[8], lrun[8];
#pragma unroll
  for (int r = 0; r < 8; ++r) { mrun[r] = -1e30f; lrun[r] = 0.f; }

  int qrow0 = qt * 128 + w * 16;
  int nkb = (qt + 1) * 2;            // 64-key blocks up to causal frontier
  int kkey = tid >> 2;               // Ks: key 0..63
  int kdbs = (tid & 3) * 16;         // Ks: d base (16 d's, paired)
  int vkp  = tid >> 3;               // Vs: key pair 0..31
  int vdbs = (tid & 7) * 8;          // Vs: 8-wide d group

  for (int kbi = 0; kbi < nkb; ++kbi) {
    int kb = kbi * 64;
    __syncthreads();
    { // stage K block into B-frag layout (k-dim = d): h2v copies
      const _Float16* src =
          Kr + ((size_t)(b * Tc + kb + kkey) * Hc + hd) * DHc + kdbs;
#pragma unroll
      for (int jj = 0; jj < 8; ++jj) {
        int d = kdbs + 2 * jj;
        h2v t = *(const h2v*)(src + 2 * jj);
        int addr = ((((d >> 5) * 4 + (kkey >> 4)) * 32) + (((d >> 4) & 1) * 16) + (kkey & 15)) * 16 + (d & 15);
        *(h2v*)(&Ks[addr]) = t;
      }
    }
    { // stage V block into B-frag layout (k-dim = key): key-pair packed h2v stores
      int key = 2 * vkp;
      const _Float16* s0 = V + ((size_t)(b * Tc + kb + key) * Hc + hd) * DHc + vdbs;
      const _Float16* s1 = s0 + (size_t)Hc * DHc;
#pragma unroll
      for (int j = 0; j < 8; ++j) {
        int d = vdbs + j;
        h2v t; t[0] = s0[j]; t[1] = s1[j];
        int addr = ((((key >> 5) * 4 + (d >> 4)) * 32) + (((key >> 4) & 1) * 16) + (d & 15)) * 16 + (key & 15);
        *(h2v*)(&Vs[addr]) = t;
      }
    }
    __syncthreads();

    // S = Q Kr^T / sqrt(DH), causal-masked
    v8f sc[4];
#pragma unroll
    for (int sub = 0; sub < 4; ++sub) {
      v16h b0 = lds_bfrag(&Ks[(0 * 4 + sub) * 512]);
      v16h b1 = lds_bfrag(&Ks[(1 * 4 + sub) * 512]);
      v8f c = __builtin_amdgcn_wmma_f32_16x16x32_f16(false, qa0, false, b0,
                                                     (short)0, zero, false, false);
      c = __builtin_amdgcn_wmma_f32_16x16x32_f16(false, qa1, false, b1,
                                                 (short)0, c, false, false);
      sc[sub] = c;
    }
#pragma unroll
    for (int sub = 0; sub < 4; ++sub)
#pragma unroll
      for (int r = 0; r < 8; ++r) {
        int grow = qrow0 + r + hi * 8;
        int gcol = kb + sub * 16 + lcol;
        float s = sc[sub][r] * 0.125f;
        if (gcol > grow) s = -1e30f;
        sc[sub][r] = s;
      }

    // online softmax (row stats via width-16 shuffles; wave32)
    float newm[8], scl[8];
#pragma unroll
    for (int r = 0; r < 8; ++r) {
      float mx = fmaxf(fmaxf(sc[0][r], sc[1][r]), fmaxf(sc[2][r], sc[3][r]));
#pragma unroll
      for (int off = 1; off < 16; off <<= 1) mx = fmaxf(mx, __shfl_xor(mx, off, 16));
      float nm = fmaxf(mrun[r], mx);
      scl[r] = __expf(mrun[r] - nm);
      newm[r] = nm;
    }
#pragma unroll
    for (int r = 0; r < 8; ++r) {
      float rs = 0.f;
#pragma unroll
      for (int sub = 0; sub < 4; ++sub) {
        float p = __expf(sc[sub][r] - newm[r]);
        sc[sub][r] = p;
        rs += p;
      }
#pragma unroll
      for (int off = 1; off < 16; off <<= 1) rs += __shfl_xor(rs, off, 16);
      lrun[r] = lrun[r] * scl[r] + rs;
      mrun[r] = newm[r];
#pragma unroll
      for (int j = 0; j < 4; ++j) ov[j][r] *= scl[r];
    }

    // P: C-layout -> A-layout via per-wave LDS bounce
#pragma unroll
    for (int sub = 0; sub < 4; ++sub)
#pragma unroll
      for (int r = 0; r < 8; ++r)
        Ps[(w * 16 + r + hi * 8) * LQ + sub * 16 + lcol] = (_Float16)sc[sub][r];

    v16h pa0 = lds_a_frag(&Ps[(w * 16) * LQ + 0], LQ);
    v16h pa1 = lds_a_frag(&Ps[(w * 16) * LQ + 32], LQ);
#pragma unroll
    for (int j = 0; j < 4; ++j) {
      v16h vb0 = lds_bfrag(&Vs[(0 * 4 + j) * 512]);
      v16h vb1 = lds_bfrag(&Vs[(1 * 4 + j) * 512]);
      ov[j] = __builtin_amdgcn_wmma_f32_16x16x32_f16(false, pa0, false, vb0,
                                                     (short)0, ov[j], false, false);
      ov[j] = __builtin_amdgcn_wmma_f32_16x16x32_f16(false, pa1, false, vb1,
                                                     (short)0, ov[j], false, false);
    }
  }

#pragma unroll
  for (int r = 0; r < 8; ++r) {
    float inv = 1.f / lrun[r];
    int grow = qrow0 + r + hi * 8;
    size_t gb = ((size_t)(b * Tc + grow) * Hc + hd) * DHc;
#pragma unroll
    for (int j = 0; j < 4; ++j) O[gb + j * 16 + lcol] = (_Float16)(ov[j][r] * inv);
  }
}

// ---------------- host launch ----------------
extern "C" void kernel_launch(void* const* d_in, const int* in_sizes, int n_in,
                              void* d_out, int out_size, void* d_ws, size_t ws_size,
                              hipStream_t stream) {
  (void)in_sizes; (void)n_in; (void)out_size; (void)ws_size;
  const int*   actions      = (const int*)d_in[0];
  const int*   observations = (const int*)d_in[1];
  const float* action_emb   = (const float*)d_in[2];
  const float* obs_emb      = (const float*)d_in[3];
  const float* lie_w        = (const float*)d_in[4];
  const float* lie_b        = (const float*)d_in[5];
  const float* Wq = (const float*)d_in[6];   const float* bq = (const float*)d_in[7];
  const float* Wk = (const float*)d_in[8];   const float* bk = (const float*)d_in[9];
  const float* Wv = (const float*)d_in[10];  const float* bv = (const float*)d_in[11];
  const float* Wo = (const float*)d_in[12];  const float* bo = (const float*)d_in[13];
  const float* ln1_g = (const float*)d_in[14]; const float* ln1_b = (const float*)d_in[15];
  const float* ln2_g = (const float*)d_in[16]; const float* ln2_b = (const float*)d_in[17];
  const float* w1 = (const float*)d_in[18];  const float* b1 = (const float*)d_in[19];
  const float* w2 = (const float*)d_in[20];  const float* b2 = (const float*)d_in[21];
  const float* out_g  = (const float*)d_in[22];
  const float* out_bl = (const float*)d_in[23];
  const float* out_w  = (const float*)d_in[24];
  const float* out_b  = (const float*)d_in[25];
  float* out = (float*)d_out;

  const size_t MT = (size_t)Bc * Tc;   // 2048 tokens

  // workspace carve-up
  char* p = (char*)d_ws;
  auto alloc = [&](size_t bytes) { char* r = p; p += (bytes + 255) & ~(size_t)255; return r; };
  float*    x    = (float*)alloc(MT * Dc * 4);
  float*    dlt  = (float*)alloc(MT * Rc * 4);
  float*    cosT = (float*)alloc(MT * Rc * 4);
  float*    sinT = (float*)alloc(MT * Rc * 4);
  _Float16* h    = (_Float16*)alloc(MT * Dc * 2);
  _Float16* q    = (_Float16*)alloc(MT * Dc * 2);
  _Float16* kbuf = (_Float16*)alloc(MT * Dc * 2);
  _Float16* v    = (_Float16*)alloc(MT * Dc * 2);
  _Float16* ao   = (_Float16*)alloc(MT * Dc * 2);
  _Float16* ffn  = (_Float16*)alloc(MT * FFc * 2);
  _Float16* wqs  = (_Float16*)alloc((size_t)Lc * Dc * Dc * 2);
  _Float16* wks  = (_Float16*)alloc((size_t)Lc * Dc * Dc * 2);
  _Float16* wvs  = (_Float16*)alloc((size_t)Lc * Dc * Dc * 2);
  _Float16* wos  = (_Float16*)alloc((size_t)Lc * Dc * Dc * 2);
  _Float16* w1s  = (_Float16*)alloc((size_t)Lc * Dc * FFc * 2);
  _Float16* w2s  = (_Float16*)alloc((size_t)Lc * FFc * Dc * 2);
  _Float16* wouts= (_Float16*)alloc((size_t)Dc * OVc * 2);

  // one-shot weight pre-swizzle (f32 -> f16 WMMA-B panel layout)
  dim3 sDD(Dc / 128, Dc / 32);
  dim3 sDF(FFc / 128, Dc / 32);
  dim3 sFD(Dc / 128, FFc / 32);
  dim3 sDO(OVc / 128, Dc / 32);
  for (int l = 0; l < Lc; ++l) {
    swz_kernel<<<sDD, 256, 0, stream>>>(Wq + (size_t)l * Dc * Dc, wqs + (size_t)l * Dc * Dc, Dc, Dc);
    swz_kernel<<<sDD, 256, 0, stream>>>(Wk + (size_t)l * Dc * Dc, wks + (size_t)l * Dc * Dc, Dc, Dc);
    swz_kernel<<<sDD, 256, 0, stream>>>(Wv + (size_t)l * Dc * Dc, wvs + (size_t)l * Dc * Dc, Dc, Dc);
    swz_kernel<<<sDD, 256, 0, stream>>>(Wo + (size_t)l * Dc * Dc, wos + (size_t)l * Dc * Dc, Dc, Dc);
    swz_kernel<<<sDF, 256, 0, stream>>>(w1 + (size_t)l * Dc * FFc, w1s + (size_t)l * Dc * FFc, FFc, Dc);
    swz_kernel<<<sFD, 256, 0, stream>>>(w2 + (size_t)l * FFc * Dc, w2s + (size_t)l * FFc * Dc, Dc, FFc);
  }
  swz_kernel<<<sDO, 256, 0, stream>>>(out_w, wouts, OVc, Dc);

  // position state: delta -> cumulative angles -> cos/sin tables
  delta_kernel<<<(int)MT, 128, 0, stream>>>(actions, action_emb, lie_w, lie_b, dlt);
  scan_kernel<<<1, 64, 0, stream>>>(dlt, cosT, sinT);
  gather_obs_kernel<<<(int)(MT * Dc / 256), 256, 0, stream>>>(observations, obs_emb, x);

  dim3 gD(Dc / 128, (int)MT / 128);
  dim3 gF(FFc / 128, (int)MT / 128);
  dim3 gO(OVc / 128, (int)MT / 128);
  const int ropeGrid = (int)(MT * Hc * Rc / 256);

  for (int l = 0; l < Lc; ++l) {
    ln_kernel<<<(int)MT, 256, 0, stream>>>(x, ln1_g + (size_t)l * Dc, ln1_b + (size_t)l * Dc, h);

    gemm_wmma_kernel<0, _Float16><<<gD, 256, 0, stream>>>(
        h, wqs + (size_t)l * Dc * Dc, bq + (size_t)l * Dc, nullptr, q, (int)MT, Dc, Dc);
    gemm_wmma_kernel<0, _Float16><<<gD, 256, 0, stream>>>(
        h, wks + (size_t)l * Dc * Dc, bk + (size_t)l * Dc, nullptr, kbuf, (int)MT, Dc, Dc);
    gemm_wmma_kernel<0, _Float16><<<gD, 256, 0, stream>>>(
        h, wvs + (size_t)l * Dc * Dc, bv + (size_t)l * Dc, nullptr, v, (int)MT, Dc, Dc);

    rope_kernel<<<ropeGrid, 256, 0, stream>>>(q,    cosT, sinT);
    rope_kernel<<<ropeGrid, 256, 0, stream>>>(kbuf, cosT, sinT);

    attn_kernel<<<Bc * Hc * (Tc / 128), 256, 0, stream>>>(q, kbuf, v, ao);

    // x += ao @ Wo + bo
    gemm_wmma_kernel<1, float><<<gD, 256, 0, stream>>>(
        ao, wos + (size_t)l * Dc * Dc, bo + (size_t)l * Dc, x, x, (int)MT, Dc, Dc);

    ln_kernel<<<(int)MT, 256, 0, stream>>>(x, ln2_g + (size_t)l * Dc, ln2_b + (size_t)l * Dc, h);

    gemm_wmma_kernel<2, _Float16><<<gF, 256, 0, stream>>>(
        h, w1s + (size_t)l * Dc * FFc, b1 + (size_t)l * FFc, nullptr, ffn, (int)MT, FFc, Dc);
    gemm_wmma_kernel<1, float><<<gD, 256, 0, stream>>>(
        ffn, w2s + (size_t)l * FFc * Dc, b2 + (size_t)l * Dc, x, x, (int)MT, Dc, FFc);
  }

  ln_kernel<<<(int)MT, 256, 0, stream>>>(x, out_g, out_bl, h);
  gemm_wmma_kernel<0, float><<<gO, 256, 0, stream>>>(
      h, wouts, out_b, nullptr, out, (int)MT, OVc, Dc);
}
